// QuantAttentionFused__68427418960039
// MI455X (gfx1250) — matrix operand order repaired and checked by reading, not compiled, not run on hardware
//
#include <hip/hip_runtime.h>
#include <hip/hip_bf16.h>

// ---------------------------------------------------------------------------
// Fused causal attention block for MI455X (gfx1250, wave32, WMMA).
//   B=2, S=2048, H=2048, NH=16, HD=128.
// Pipeline:
//   0) fp32 -> bf16 conversion pre-pass (bandwidth bound, keeps conversions
//      OUT of the matmul inner loops)
//   1) QKV projection  (bf16 WMMA, fp32 accum), writes Q,K [bh][s][d], V^T
//   2) flash attention (bf16 WMMA QK^T and PV, fp32 online softmax)
//   3) output projection (bf16 WMMA), fp32 result
// ---------------------------------------------------------------------------

typedef __attribute__((ext_vector_type(4)))  float        f32x4;
typedef __attribute__((ext_vector_type(8)))  float        v8f;
typedef __attribute__((ext_vector_type(16))) __bf16       v16bf;
typedef __attribute__((ext_vector_type(4)))  unsigned int u32x4;

union BF16V {
    v16bf          v;
    unsigned short u[16];
    u32x4          h[2];
};

static __device__ __forceinline__ unsigned short f2bf(float f) {
    union { float f; unsigned int u; } x; x.f = f;
    unsigned int r = 0x7FFFu + ((x.u >> 16) & 1u);   // round-to-nearest-even
    return (unsigned short)((x.u + r) >> 16);
}

#define WMMA_BF16(A, Bv, C) \
    __builtin_amdgcn_wmma_f32_16x16x32_bf16(false, (A), false, (Bv), (short)0, (C), false, false)

static constexpr int  BN  = 2;
static constexpr int  SN  = 2048;
static constexpr int  HN  = 2048;
static constexpr int  NHN = 16;
static constexpr int  HDN = 128;
static constexpr long HS_E  = (long)BN * SN * HN;        //  8388608 hidden elems
static constexpr long WP_E  = (long)3 * HN * HN;         // 12582912 w_pack elems
static constexpr long WO_E  = (long)HN * HN;             //  4194304 w_o elems
static constexpr long QKV_E = (long)BN * NHN * SN * HDN; //  8388608 per Q/K/V

// Load a 16-wide bf16 WMMA A-operand: rows per-lane, K split {kb..kb+7, kb+16..kb+23}.
static __device__ __forceinline__ v16bf loadA(const unsigned short* p) {
    BF16V t;
    t.h[0] = *(const u32x4*)(p);
    t.h[1] = *(const u32x4*)(p + 16);
    return t.v;
}
// Load a bf16 WMMA B-operand: K split {kb..kb+7, kb+16..kb+23}.
static __device__ __forceinline__ v16bf loadB(const unsigned short* p) {
    BF16V t;
    t.h[0] = *(const u32x4*)(p);
    t.h[1] = *(const u32x4*)(p + 16);
    return t.v;
}

// ---------------------------------------------------------------------------
// Kernel 0: fp32 -> bf16 bulk conversion, 8 elements per thread.
// ---------------------------------------------------------------------------
__global__ void __launch_bounds__(256)
cvt_kernel(const float* __restrict__ src, unsigned short* __restrict__ dst, int n8)
{
    const int i = blockIdx.x * 256 + threadIdx.x;
    if (i >= n8) return;
    const f32x4* s = (const f32x4*)src + (long)i * 2;
    f32x4 a = s[0], b = s[1];
    union { u32x4 v; unsigned short u[8]; } o;
#pragma unroll
    for (int k = 0; k < 4; ++k) { o.u[k] = f2bf(a[k]); o.u[4 + k] = f2bf(b[k]); }
    ((u32x4*)dst)[i] = o.v;
}

// ---------------------------------------------------------------------------
// Kernel 1: fused QKV projection (bf16 GEMM).
//   A = hidden_bf16 [4096 x 2048], W = w_pack_bf16 [6144 x 2048], C = A*W^T.
// Each wave: 32(M) x 64(N) tile -> 8 independent WMMA accum chains.
// Q gets 1/sqrt(128) folded in; V is stored transposed [bh][d][s].
// ---------------------------------------------------------------------------
__global__ void __launch_bounds__(128)
qkv_kernel(const unsigned short* __restrict__ hb, const unsigned short* __restrict__ wpb,
           unsigned short* __restrict__ qs, unsigned short* __restrict__ ks,
           unsigned short* __restrict__ vt)
{
    const int lane  = threadIdx.x & 31;
    const int wave  = blockIdx.x * 4 + (threadIdx.x >> 5);
    const int mPair = wave & 127;           // 128 pairs of 16-row tiles (32 rows)
    const int nChunk = wave >> 7;           // 96 N chunks of 64
    const int half  = lane >> 4;
    const int ln    = lane & 15;

    const unsigned short* A0 = hb + (long)(mPair * 32 + ln) * HN;
    const unsigned short* A1 = A0 + (long)16 * HN;
    const unsigned short* Wr[4];
#pragma unroll
    for (int t = 0; t < 4; ++t)
        Wr[t] = wpb + (long)(nChunk * 64 + t * 16 + ln) * HN;

    v8f c0[4] = {}, c1[4] = {};
    for (int kk = 0; kk < HN; kk += 32) {
        const int ka = kk + half * 8;
        const int kb = kk + half * 8;
        __builtin_prefetch(A0 + ka + 128, 0, 1);
        v16bf a0 = loadA(A0 + ka);
        v16bf a1 = loadA(A1 + ka);
#pragma unroll
        for (int t = 0; t < 4; ++t) {
            v16bf Bv = loadB(Wr[t] + kb);
            c0[t] = WMMA_BF16(a0, Bv, c0[t]);
            c1[t] = WMMA_BF16(a1, Bv, c1[t]);
        }
    }

    const float qscale = 0.08838834764831845f;  // 1/sqrt(128)
#pragma unroll
    for (int t = 0; t < 4; ++t) {
        const int o     = nChunk * 64 + t * 16 + ln;
        const int which = o >> 11;              // 0=q 1=k 2=v (uniform per t)
        const int hh    = o & 2047;
        const int head  = hh >> 7;
        const int d     = hh & 127;
#pragma unroll
        for (int p = 0; p < 2; ++p) {
#pragma unroll
            for (int r = 0; r < 8; ++r) {
                const int m = mPair * 32 + p * 16 + r + half * 8;
                const int b = m >> 11, s = m & 2047;
                const int bh = b * NHN + head;
                const float v = p ? c1[t][r] : c0[t][r];
                if (which == 0)      qs[((long)bh * SN + s) * HDN + d] = f2bf(v * qscale);
                else if (which == 1) ks[((long)bh * SN + s) * HDN + d] = f2bf(v);
                else                 vt[((long)bh * HDN + d) * SN + s] = f2bf(v);
            }
        }
    }
}

// ---------------------------------------------------------------------------
// Kernel 2: flash attention, one wave = 16 query rows of one (b, head).
// ---------------------------------------------------------------------------
__global__ void __launch_bounds__(128)
attn_kernel(const unsigned short* __restrict__ qs, const unsigned short* __restrict__ ks,
            const unsigned short* __restrict__ vt, unsigned short* __restrict__ aob)
{
    __shared__ __align__(16) unsigned short plds[4 * 16 * 32];   // 1KB per wave

    const int lane = threadIdx.x & 31;
    const int wib  = threadIdx.x >> 5;
    const int wave = blockIdx.x * 4 + wib;       // 4096 waves
    const int q0   = (wave & 127) * 16;
    const int bh   = wave >> 7;                  // 0..31
    const int b    = bh >> 4, head = bh & 15;
    const int half = lane >> 4, ln = lane & 15;

    const unsigned short* Q = qs + (long)bh * SN * HDN;
    const unsigned short* K = ks + (long)bh * SN * HDN;
    const unsigned short* V = vt + (long)bh * HDN * SN;

    v16bf qa[4];
#pragma unroll
    for (int cc = 0; cc < 4; ++cc)
        qa[cc] = loadA(Q + (long)(q0 + ln) * HDN + cc * 32 + half * 8);

    v8f   O[8] = {};
    float mrow[8], lrow[8];
#pragma unroll
    for (int r = 0; r < 8; ++r) { mrow[r] = -1e30f; lrow[r] = 0.0f; }

    unsigned short* lds = plds + wib * 512;
    const int nCh = ((q0 + 15) >> 5) + 1;        // causal: kv chunks of 32

    for (int j = 0; j < nCh; ++j) {
        const int kv0 = j * 32;

        // ---- scores: two 16-col tiles, 4 WMMAs each over HD=128 ----
        v8f s0 = {}, s1 = {};
        const unsigned short* kp0 = K + (long)(kv0 + ln) * HDN + half * 8;
        const unsigned short* kp1 = kp0 + (long)16 * HDN;
#pragma unroll
        for (int cc = 0; cc < 4; ++cc) {
            s0 = WMMA_BF16(qa[cc], loadB(kp0 + cc * 32), s0);
            s1 = WMMA_BF16(qa[cc], loadB(kp1 + cc * 32), s1);
        }

        // ---- causal mask + online softmax update ----
#pragma unroll
        for (int r = 0; r < 8; ++r) {
            const int rowa = q0 + r + half * 8;
            const int col0 = kv0 + ln, col1 = col0 + 16;
            float x0 = (col0 > rowa) ? -1e30f : s0[r];
            float x1 = (col1 > rowa) ? -1e30f : s1[r];

            float tm = fmaxf(x0, x1);
#pragma unroll
            for (int mk = 1; mk < 16; mk <<= 1) tm = fmaxf(tm, __shfl_xor(tm, mk, 32));
            const float mn = fmaxf(mrow[r], tm);
            const float sc = __expf(mrow[r] - mn);
            mrow[r] = mn;

            const float p0 = __expf(x0 - mn);
            const float p1 = __expf(x1 - mn);
            float rs = p0 + p1;
#pragma unroll
            for (int mk = 1; mk < 16; mk <<= 1) rs += __shfl_xor(rs, mk, 32);
            lrow[r] = lrow[r] * sc + rs;

#pragma unroll
            for (int t = 0; t < 8; ++t) O[t][r] *= sc;

            const int lr = r + half * 8;          // C-layout row -> LDS row-major
            lds[lr * 32 + ln]      = f2bf(p0);
            lds[lr * 32 + ln + 16] = f2bf(p1);
        }

        // Same-wave LDS ops are in-order (DScnt); explicit fence for safety.
        asm volatile("s_wait_dscnt 0" ::: "memory");

        // ---- reload P as 16x32 A-operand, then O += P * V ----
        v16bf pa = loadA(lds + ln * 32 + half * 8);
#pragma unroll
        for (int t = 0; t < 8; ++t) {
            const unsigned short* vp = V + (long)(t * 16 + ln) * SN + kv0 + half * 8;
            O[t] = WMMA_BF16(pa, loadB(vp), O[t]);
        }
    }

    // ---- epilogue: normalize, write bf16 [b][s][h] for the next GEMM ----
#pragma unroll
    for (int r = 0; r < 8; ++r) {
        const float inv = 1.0f / lrow[r];
        const int   s   = q0 + r + half * 8;
        unsigned short* orow = aob + ((long)(b * SN + s)) * HN + head * HDN;
#pragma unroll
        for (int t = 0; t < 8; ++t) orow[t * 16 + ln] = f2bf(O[t][r] * inv);
    }
}

// ---------------------------------------------------------------------------
// Kernel 3: output projection.  out_f32 = aob [4096x2048] * w_o_bf16^T.
// Each wave: 32(M) x 64(N) tile.
// ---------------------------------------------------------------------------
__global__ void __launch_bounds__(128)
oproj_kernel(const unsigned short* __restrict__ aob, const unsigned short* __restrict__ wob,
             float* __restrict__ out)
{
    const int lane  = threadIdx.x & 31;
    const int wave  = blockIdx.x * 4 + (threadIdx.x >> 5);
    const int mPair = wave & 127;           // 128 pairs (32 rows)
    const int nChunk = wave >> 7;           // 32 N chunks of 64
    const int half  = lane >> 4;
    const int ln    = lane & 15;

    const unsigned short* A0 = aob + (long)(mPair * 32 + ln) * HN;
    const unsigned short* A1 = A0 + (long)16 * HN;
    const unsigned short* Wr[4];
#pragma unroll
    for (int t = 0; t < 4; ++t)
        Wr[t] = wob + (long)(nChunk * 64 + t * 16 + ln) * HN;

    v8f c0[4] = {}, c1[4] = {};
    for (int kk = 0; kk < HN; kk += 32) {
        const int ka = kk + half * 8;
        const int kb = kk + half * 8;
        __builtin_prefetch(A0 + ka + 128, 0, 1);
        v16bf a0 = loadA(A0 + ka);
        v16bf a1 = loadA(A1 + ka);
#pragma unroll
        for (int t = 0; t < 4; ++t) {
            v16bf Bv = loadB(Wr[t] + kb);
            c0[t] = WMMA_BF16(a0, Bv, c0[t]);
            c1[t] = WMMA_BF16(a1, Bv, c1[t]);
        }
    }

#pragma unroll
    for (int t = 0; t < 4; ++t) {
        const int n = nChunk * 64 + t * 16 + ln;
#pragma unroll
        for (int p = 0; p < 2; ++p) {
#pragma unroll
            for (int r = 0; r < 8; ++r) {
                const int m = mPair * 32 + p * 16 + r + half * 8;
                out[(long)m * HN + n] = p ? c1[t][r] : c0[t][r];
            }
        }
    }
}

// ---------------------------------------------------------------------------
// Host launch.  Inputs: [0]=hidden f32, [1]=attention_mask (unused, causal is
// analytic), [2]=w_pack f32, [3]=w_o f32.
// Workspace (bf16): hb 16.8MB | wpb 25.2MB | wob 8.4MB | Q/K/V^T 3x16.8MB |
// attn-out 16.8MB  => ~117 MB total.
// ---------------------------------------------------------------------------
extern "C" void kernel_launch(void* const* d_in, const int* in_sizes, int n_in,
                              void* d_out, int out_size, void* d_ws, size_t ws_size,
                              hipStream_t stream)
{
    const float* hs = (const float*)d_in[0];
    const float* wp = (const float*)d_in[2];
    const float* wo = (const float*)d_in[3];
    float*       out = (float*)d_out;

    unsigned short* hb  = (unsigned short*)d_ws;
    unsigned short* wpb = hb  + HS_E;
    unsigned short* wob = wpb + WP_E;
    unsigned short* qs  = wob + WO_E;
    unsigned short* ks  = qs  + QKV_E;
    unsigned short* vt  = ks  + QKV_E;
    unsigned short* aob = vt  + QKV_E;

    // 0) bf16 conversion pre-pass (bandwidth bound)
    cvt_kernel<<<(int)(HS_E / 8 / 256), 256, 0, stream>>>(hs, hb,  (int)(HS_E / 8));
    cvt_kernel<<<(int)(WP_E / 8 / 256), 256, 0, stream>>>(wp, wpb, (int)(WP_E / 8));
    cvt_kernel<<<(int)(WO_E / 8 / 256), 256, 0, stream>>>(wo, wob, (int)(WO_E / 8));

    // 1) QKV projection: 128 M-pairs * 96 N-chunks = 12288 waves / 4 per block
    qkv_kernel<<<3072, 128, 0, stream>>>(hb, wpb, qs, ks, vt);
    // 2) attention: 32 (b,head) * 128 q-tiles = 4096 waves / 4
    attn_kernel<<<1024, 128, 0, stream>>>(qs, ks, vt, aob);
    // 3) output projection: 128 M-pairs * 32 N-chunks = 4096 waves / 4
    oproj_kernel<<<1024, 128, 0, stream>>>(aob, wob, out);
}
